// SelfAttention_13134009991901
// MI455X (gfx1250) — matrix operand order repaired and checked by reading, not compiled, not run on hardware
//
#include <hip/hip_runtime.h>

// ---- problem dims (fixed by reference) ----
#define BB   4
#define CC   256
#define CO   32
#define NN   4096
#define ROWS 320   // CO (q) + CO (k) + CC (v) rows of the fused projection

typedef float  v2f  __attribute__((ext_vector_type(2)));
typedef float  v8f  __attribute__((ext_vector_type(8)));
typedef __bf16 v8bf __attribute__((ext_vector_type(8)));
typedef __bf16 v16bf __attribute__((ext_vector_type(16)));

union BF16x16 { v16bf v; v8bf h[2]; __bf16 e[16]; };

// ---- CDNA5 async memory->LDS path (ISA §10.7 / §15.18, ASYNCcnt-tracked) ----
typedef __attribute__((address_space(3))) void lds_void;

__device__ __forceinline__ void async_copy_f32x2(const float* g, float* l) {
    // LDS byte offset: AS(3) pointers are 32-bit LDS offsets on amdgcn.
    unsigned loff = (unsigned)(unsigned long long)(lds_void*)l;
    asm volatile("global_load_async_to_lds_b64 %0, %1, off"
                 :: "v"(loff), "v"((unsigned long long)g)
                 : "memory");
}
__device__ __forceinline__ void wait_async_zero() {
    asm volatile("s_wait_asynccnt 0x0" ::: "memory");
}

// ============================================================================
// Kernel 1: fused projection  Out(320 x N) = [Wq;Wk;Wv] * x + [bq;bk;bv]
//   rows   0..31  -> Q  stored (b, c, m) row-major          (f32)
//   rows  32..63  -> K  stored transposed (b, n, c)         (f32, for A-tiles)
//   rows  64..319 -> V  stored (b, c, n) row-major          (bf16)
// fp32 WMMA 16x16x4, K-loop over C=256.
// ============================================================================
__global__ __launch_bounds__(32) void proj_kernel(
    const float* __restrict__ x,
    const float* __restrict__ Wq, const float* __restrict__ bq,
    const float* __restrict__ Wk, const float* __restrict__ bk,
    const float* __restrict__ Wv, const float* __restrict__ bv,
    float* __restrict__ qws, float* __restrict__ ktws, __bf16* __restrict__ vws)
{
    const int lane  = threadIdx.x;
    const int col16 = lane & 15;      // tile column (n)
    const int h     = lane >> 4;      // lane half
    const int n0    = blockIdx.x * 16;
    const int row0  = blockIdx.y * 16;
    const int b     = blockIdx.z;

    const float* Wsel; int lrow0;
    if      (row0 < 32) { Wsel = Wq; lrow0 = row0;      }
    else if (row0 < 64) { Wsel = Wk; lrow0 = row0 - 32; }
    else                { Wsel = Wv; lrow0 = row0 - 64; }

    const float* xb   = x + (size_t)b * CC * NN;
    // A lane view: A[row][k], row = lane%16, k-pair base = 2*h  (ISA f32 16x4 layout)
    const float* arow = Wsel + (size_t)(lrow0 + col16) * CC + 2 * h;
    // B lane view: B[k][j],  j = lane%16, k rows = 2*h, 2*h+1
    const float* bcol = xb + (size_t)(2 * h) * NN + n0 + col16;

    v8f acc = {};
    #pragma unroll 4
    for (int kc = 0; kc < CC; kc += 4) {
        v2f a = *(const v2f*)(arow + kc);
        v2f bb;
        bb.x = bcol[(size_t)kc * NN];
        bb.y = bcol[(size_t)kc * NN + NN];
        acc = __builtin_amdgcn_wmma_f32_16x16x4_f32(false, a, false, bb,
                                                    (short)0, acc, false, false);
    }

    // D element acc[r] -> row M = r + 8*h, col = n0 + col16
    if (row0 < 32) {
        #pragma unroll
        for (int r = 0; r < 8; ++r) {
            int gr = row0 + r + 8 * h;
            qws[((size_t)b * CO + gr) * NN + n0 + col16] = acc[r] + bq[gr];
        }
    } else if (row0 < 64) {
        int cbase = (row0 - 32) + 8 * h;
        float* dst = ktws + ((size_t)b * NN + n0 + col16) * CO + cbase;
        #pragma unroll
        for (int r = 0; r < 8; ++r) dst[r] = acc[r] + bk[cbase + r];
    } else {
        #pragma unroll
        for (int r = 0; r < 8; ++r) {
            int c = (row0 - 64) + r + 8 * h;
            vws[((size_t)b * CC + c) * NN + n0 + col16] = (__bf16)(acc[r] + bv[c]);
        }
    }
}

// ============================================================================
// Kernel 2: raw attention logits  S[b,n,m] = sum_c Kt[b,n,c] * Q[b,c,m]
// fp32 WMMA 16x16x4, K-loop over Co=32 (8 steps). Writes into attn slice of d_out.
// ============================================================================
__global__ __launch_bounds__(32) void scores_kernel(
    const float* __restrict__ ktws, const float* __restrict__ qws,
    float* __restrict__ sc)
{
    const int lane  = threadIdx.x;
    const int col16 = lane & 15;
    const int h     = lane >> 4;
    const int m0    = blockIdx.x * 16;
    const int n0    = blockIdx.y * 16;
    const int b     = blockIdx.z;

    const float* ktrow = ktws + ((size_t)b * NN + n0 + col16) * CO + 2 * h; // A
    const float* qcol  = qws + ((size_t)b * CO + 2 * h) * NN + m0 + col16;  // B

    v8f acc = {};
    #pragma unroll
    for (int kc = 0; kc < CO; kc += 4) {
        v2f a = *(const v2f*)(ktrow + kc);
        v2f bb;
        bb.x = qcol[(size_t)kc * NN];
        bb.y = qcol[(size_t)kc * NN + NN];
        acc = __builtin_amdgcn_wmma_f32_16x16x4_f32(false, a, false, bb,
                                                    (short)0, acc, false, false);
    }
    #pragma unroll
    for (int r = 0; r < 8; ++r)
        sc[((size_t)b * NN + n0 + r + 8 * h) * NN + m0 + col16] = acc[r];
}

// ============================================================================
// Kernel 3: numerically-stable in-place row softmax over m (4096 per row).
// 256 threads/row, 16 values/thread, LDS tree reductions.
// ============================================================================
__global__ __launch_bounds__(256) void softmax_kernel(float* __restrict__ sc)
{
    __shared__ float red[256];
    const int t = threadIdx.x;
    float* row = sc + (size_t)blockIdx.x * NN;

    float vals[16];
    float m = -3.402823466e38f;
    #pragma unroll
    for (int i = 0; i < 16; ++i) {
        vals[i] = row[t + 256 * i];
        m = fmaxf(m, vals[i]);
    }
    red[t] = m; __syncthreads();
    for (int s = 128; s > 0; s >>= 1) {
        if (t < s) red[t] = fmaxf(red[t], red[t + s]);
        __syncthreads();
    }
    m = red[0]; __syncthreads();

    float sum = 0.f;
    #pragma unroll
    for (int i = 0; i < 16; ++i) { vals[i] = __expf(vals[i] - m); sum += vals[i]; }
    red[t] = sum; __syncthreads();
    for (int s = 128; s > 0; s >>= 1) {
        if (t < s) red[t] += red[t + s];
        __syncthreads();
    }
    const float inv = 1.0f / red[0];
    #pragma unroll
    for (int i = 0; i < 16; ++i) row[t + 256 * i] = vals[i] * inv;
}

// ============================================================================
// Kernel 4: out[b,c,m] = gamma * sum_n V[b,c,n]*attn[b,n,m] + x[b,c,m]
// One 256-thread block (8 waves) owns a full 256-channel x 16-m column strip.
// Per 32-wide K-step the 32x16 attn f32 tile is streamed memory->LDS with
// GLOBAL_LOAD_ASYNC_TO_LDS_B64 into a DOUBLE-BUFFERED LDS tile (next step's
// copy overlaps this step's WMMAs; ASYNCcnt + workgroup barrier for ordering).
// Each wave builds the shared bf16 B fragment from LDS (with f32->bf16 cvt)
// and issues 2x v_wmma_f32_16x16x32_bf16 (its two c-tiles).
// attn is read from HBM exactly once.
// ============================================================================
#define SROW 18   // padded LDS row stride (dwords): lanes 16-31 avoid the
                  // banks used by lanes 0-15 on the strided fragment reads

__global__ __launch_bounds__(256) void av_kernel(
    const __bf16* __restrict__ vws, const float* __restrict__ attn,
    const float* __restrict__ x, const float* __restrict__ gamma,
    float* __restrict__ out)
{
    __shared__ float stile[2][32 * SROW];   // double-buffered 32(n) x 16(m) f32 tile

    const int t     = threadIdx.x;
    const int lane  = t & 31;
    const int wave  = t >> 5;          // 0..7
    const int col16 = lane & 15;
    const int h     = lane >> 4;
    const int m0    = blockIdx.x * 16;
    const int b     = blockIdx.z;

    const int c0a = wave * 32;         // this wave's two c-tiles
    const int c0b = c0a + 16;

    const __bf16* vrowA = vws + (size_t)(b * CC + c0a + col16) * NN;
    const __bf16* vrowB = vws + (size_t)(b * CC + c0b + col16) * NN;

    // staging map: thread t -> n-row = t>>3, m-pair = (t&7)*2 (coalesced b64)
    const int sn  = t >> 3;
    const int smp = (t & 7) * 2;
    const float* stage = attn + ((size_t)b * NN + sn) * NN + m0 + smp;
    float* sl0 = &stile[0][sn * SROW + smp];
    float* sl1 = &stile[1][sn * SROW + smp];

    // prologue: stream tile 0 into buffer 0
    async_copy_f32x2(stage, sl0);

    v8f acc0 = {}, acc1 = {};
    int cur = 0;
    for (int nb = 0; nb < NN; nb += 32) {
        wait_async_zero();   // this wave's async LDS writes have landed
        __syncthreads();     // publish across all 8 waves; fence prior reads
        if (nb + 32 < NN)    // stream NEXT tile into the other buffer
            async_copy_f32x2(stage + (size_t)(nb + 32) * NN, cur ? sl0 : sl1);

        // bf16 B 32x16 lane layout: e<8 -> K=8h+e ; e>=8 -> K=16+8h+(e-8)
        const float* sb = stile[cur];
        BF16x16 bbv;
        #pragma unroll
        for (int e = 0; e < 16; ++e)
            bbv.e[e] = (__bf16)sb[(e + 8 * h + (e >= 8 ? 8 : 0)) * SROW + col16];

        // bf16 A 16x32 lane layout: e<8 -> K=8h+e ; e>=8 -> K=16+8h+(e-8)
        BF16x16 aa;
        aa.h[0] = *(const v8bf*)(vrowA + nb + 8 * h);
        aa.h[1] = *(const v8bf*)(vrowA + nb + 16 + 8 * h);
        acc0 = __builtin_amdgcn_wmma_f32_16x16x32_bf16(false, aa.v, false, bbv.v,
                                                       (short)0, acc0, false, false);
        aa.h[0] = *(const v8bf*)(vrowB + nb + 8 * h);
        aa.h[1] = *(const v8bf*)(vrowB + nb + 16 + 8 * h);
        acc1 = __builtin_amdgcn_wmma_f32_16x16x32_bf16(false, aa.v, false, bbv.v,
                                                       (short)0, acc1, false, false);
        cur ^= 1;
    }

    const float g = gamma[0];
    #pragma unroll
    for (int r = 0; r < 8; ++r) {
        int c = c0a + r + 8 * h;
        size_t idx = ((size_t)b * CC + c) * NN + m0 + col16;
        out[idx] = g * acc0[r] + x[idx];
    }
    #pragma unroll
    for (int r = 0; r < 8; ++r) {
        int c = c0b + r + 8 * h;
        size_t idx = ((size_t)b * CC + c) * NN + m0 + col16;
        out[idx] = g * acc1[r] + x[idx];
    }
}

// ============================================================================
extern "C" void kernel_launch(void* const* d_in, const int* in_sizes, int n_in,
                              void* d_out, int out_size, void* d_ws, size_t ws_size,
                              hipStream_t stream) {
    (void)in_sizes; (void)n_in; (void)out_size; (void)ws_size;
    const float* x     = (const float*)d_in[0];
    const float* Wq    = (const float*)d_in[1];
    const float* bq    = (const float*)d_in[2];
    const float* Wk    = (const float*)d_in[3];
    const float* bk    = (const float*)d_in[4];
    const float* Wv    = (const float*)d_in[5];
    const float* bv    = (const float*)d_in[6];
    const float* gamma = (const float*)d_in[7];

    float* out  = (float*)d_out;
    float* attn = out + (size_t)BB * CC * NN;   // outputs concatenated: out, attn

    // workspace: Q (2MB f32) | Kt (2MB f32) | V (8.4MB bf16)  ~= 12.6 MB
    float*  qws  = (float*)d_ws;
    float*  ktws = qws + (size_t)BB * CO * NN;
    __bf16* vws  = (__bf16*)(ktws + (size_t)BB * NN * CO);

    proj_kernel<<<dim3(NN / 16, ROWS / 16, BB), 32, 0, stream>>>(
        x, Wq, bq, Wk, bk, Wv, bv, qws, ktws, vws);
    scores_kernel<<<dim3(NN / 16, NN / 16, BB), 32, 0, stream>>>(ktws, qws, attn);
    softmax_kernel<<<dim3(BB * NN), 256, 0, stream>>>(attn);
    av_kernel<<<dim3(NN / 16, 1, BB), 256, 0, stream>>>(vws, attn, x, gamma, out);
}
